// ContinuousDiscreteDEformer_33122787786997
// MI455X (gfx1250) — matrix operand order repaired and checked by reading, not compiled, not run on hardware
//
#include <hip/hip_runtime.h>
#include <math.h>

typedef _Float16 v16h __attribute__((ext_vector_type(16)));
typedef _Float16 v8h  __attribute__((ext_vector_type(8)));
typedef float    v8f  __attribute__((ext_vector_type(8)));

#define Bsz 256
#define Dd  128
#define Hh  128
#define Tt  256   // 2*Dd
#define NHh 4
#define DH  32

static __device__ __forceinline__ v8f wmma_f16(v16h a, v16h b, v8f c) {
  return __builtin_amdgcn_wmma_f32_16x16x32_f16(false, a, false, b, (short)0, c, false, false);
}

static __device__ __forceinline__ float gelu_tanh(float v) {
  float c = 0.7978845608028654f * (v + 0.044715f * v * v * v);
  return 0.5f * v * (1.0f + tanhf(c));
}

// ---------------------------------------------------------------------------
// Weight convert+transpose: W (K x N) f32 row-major  ->  WT (N x K) f16
// ---------------------------------------------------------------------------
__global__ void convT_kernel(const float* __restrict__ w, _Float16* __restrict__ wt,
                             int K, int N) {
  int i = blockIdx.x * 256 + threadIdx.x;
  if (i >= K * N) return;
  int k = i / N, n = i % N;
  wt[(size_t)n * K + k] = (_Float16)w[i];
}

// ---------------------------------------------------------------------------
// inv[b, order[b,j]] = j
// ---------------------------------------------------------------------------
__global__ void inv_kernel(const int* __restrict__ order, int* __restrict__ inv) {
  int b = blockIdx.x, j = threadIdx.x;
  inv[b * Dd + order[b * Dd + j]] = j;
}

// ---------------------------------------------------------------------------
// idxfeat_table[d] = mlp_index(index_emb[d]) * sqrt(H)   (128 rows)
// ---------------------------------------------------------------------------
__global__ __launch_bounds__(128) void idx_table_kernel(
    const float* __restrict__ emb,
    const float* __restrict__ w1, const float* __restrict__ b1,
    const float* __restrict__ w2, const float* __restrict__ b2,
    float* __restrict__ out) {
  int row = blockIdx.x; int t = threadIdx.x;
  __shared__ float sin_[32];
  __shared__ float sh[128];
  if (t < 32) sin_[t] = emb[row * 32 + t];
  __syncthreads();
  float a = b1[t];
  for (int k = 0; k < 32; k++) a += sin_[k] * w1[k * 128 + t];
  sh[t] = fmaxf(a, 0.f);
  __syncthreads();
  float o = b2[t];
  for (int k = 0; k < 128; k++) o += sh[k] * w2[k * 128 + t];
  out[row * 128 + t] = o * 11.313708498984761f;
}

// ---------------------------------------------------------------------------
// discf_table[kf*10+c] = mlp_disc(concat(index_emb[64+kf], class_emb[kf*10+c])) * sqrt(H)
// ---------------------------------------------------------------------------
__global__ __launch_bounds__(128) void disc_table_kernel(
    const float* __restrict__ iemb, const float* __restrict__ cemb,
    const float* __restrict__ w1, const float* __restrict__ b1,
    const float* __restrict__ w2, const float* __restrict__ b2,
    float* __restrict__ out) {
  int row = blockIdx.x;            // 0..639
  int kf = row / 10, c = row % 10;
  int t = threadIdx.x;
  __shared__ float sin_[64];
  __shared__ float sh[128];
  if (t < 32)      sin_[t] = iemb[(64 + kf) * 32 + t];
  else if (t < 64) sin_[t] = cemb[(kf * 10 + c) * 32 + (t - 32)];
  __syncthreads();
  float a = b1[t];
  for (int k = 0; k < 64; k++) a += sin_[k] * w1[k * 128 + t];
  sh[t] = fmaxf(a, 0.f);
  __syncthreads();
  float o = b2[t];
  for (int k = 0; k < 128; k++) o += sh[k] * w2[k * 128 + t];
  out[row * 128 + t] = o * 11.313708498984761f;
}

// ---------------------------------------------------------------------------
// cont_f[b,kf] = mlp_cont(concat(index_emb[kf], x[b,kf])) * sqrt(H)
// ---------------------------------------------------------------------------
__global__ __launch_bounds__(128) void cont_feat_kernel(
    const float* __restrict__ iemb, const float* __restrict__ x,
    const float* __restrict__ w1, const float* __restrict__ b1,
    const float* __restrict__ w2, const float* __restrict__ b2,
    float* __restrict__ out) {
  int row = blockIdx.x;            // b*64 + kf
  int b = row >> 6, kf = row & 63;
  int t = threadIdx.x;
  __shared__ float sin_[33];
  __shared__ float sh[128];
  if (t < 32) sin_[t] = iemb[kf * 32 + t];
  if (t == 32) sin_[32] = x[b * Dd + kf];
  __syncthreads();
  float a = b1[t];
  for (int k = 0; k < 33; k++) a += sin_[k] * w1[k * 128 + t];
  sh[t] = fmaxf(a, 0.f);
  __syncthreads();
  float o = b2[t];
  for (int k = 0; k < 128; k++) o += sh[k] * w2[k * 128 + t];
  out[(size_t)row * 128 + t] = o * 11.313708498984761f;
}

// ---------------------------------------------------------------------------
// h0[b, 2j]   = idxfeat_table[order[b,j]]
// h0[b, 2j+1] = f<64 ? cont_f[b,f] : discf_table[(f-64)*10 + class]
// ---------------------------------------------------------------------------
__global__ __launch_bounds__(128) void assemble_kernel(
    const int* __restrict__ order, const float* __restrict__ x,
    const float* __restrict__ idxT, const float* __restrict__ discT,
    const float* __restrict__ contf, float* __restrict__ h) {
  int bj = blockIdx.x; int b = bj >> 7; int j = bj & 127;
  int t = threadIdx.x;
  int f = order[b * Dd + j];
  size_t base = ((size_t)b * Tt + 2 * j) * Hh;
  h[base + t] = idxT[f * 128 + t];
  float val;
  if (f < 64) {
    val = contf[((size_t)b * 64 + f) * 128 + t];
  } else {
    int c = (int)x[b * Dd + f];
    val = discT[((f - 64) * 10 + c) * 128 + t];
  }
  h[base + Hh + t] = val;
}

// ---------------------------------------------------------------------------
// LayerNorm over H=128, output f16
// ---------------------------------------------------------------------------
__global__ __launch_bounds__(128) void ln_kernel(
    const float* __restrict__ h, const float* __restrict__ g,
    const float* __restrict__ bb, _Float16* __restrict__ out) {
  int token = blockIdx.x; int t = threadIdx.x;
  const float* row = h + (size_t)token * Hh;
  float v = row[t];
  __shared__ float red[128];
  red[t] = v; __syncthreads();
  for (int s = 64; s > 0; s >>= 1) { if (t < s) red[t] += red[t + s]; __syncthreads(); }
  float mu = red[0] * (1.0f / 128.0f);
  __syncthreads();
  float d = v - mu;
  red[t] = d * d; __syncthreads();
  for (int s = 64; s > 0; s >>= 1) { if (t < s) red[t] += red[t + s]; __syncthreads(); }
  float var = red[0] * (1.0f / 128.0f);
  float r = rsqrtf(var + 1e-5f);
  out[(size_t)token * Hh + t] = (_Float16)(d * r * g[t] + bb[t]);
}

// ---------------------------------------------------------------------------
// WMMA GEMM: C[M,N] = act(A[M,K] @ W + bias), W stored transposed f16 (N x K)
// One wave computes a 16x64 output tile.  ACT: 0=none, 1=gelu.
// ADDF: 0 -> outH f16 store; 1 -> outF f32 accumulate (residual add)
// ---------------------------------------------------------------------------
template <int ACT, int ADDF>
__global__ __launch_bounds__(32) void gemm_wmma_kernel(
    const _Float16* __restrict__ A, const _Float16* __restrict__ WT,
    const float* __restrict__ bias, _Float16* __restrict__ outH,
    float* __restrict__ outF, int K, int N) {
  const int m0 = blockIdx.x * 16;
  const int n0 = blockIdx.y * 64;
  const int lane = threadIdx.x;
  const int hf = lane >> 4;        // which 16-lane half
  const int lm = lane & 15;
  v8f acc[4] = {};
  const _Float16* arow = A + (size_t)(m0 + lm) * K + hf * 16;
  for (int kk = 0; kk < K; kk += 32) {
    // A fragment 16x32: halves 0-15 -> K=kk+hf*16..kk+hf*16+15 (contiguous)
    v16h af;
    v8h lo = *(const v8h*)(arow + kk);
    v8h hi = *(const v8h*)(arow + kk + 8);
#pragma unroll
    for (int i = 0; i < 8; i++) { af[i] = lo[i]; af[8 + i] = hi[i]; }
#pragma unroll
    for (int t = 0; t < 4; t++) {
      // B fragment 32x16: lane=column, halves -> K=kk+hf*16..kk+hf*16+15 (contiguous in WT)
      const _Float16* wp = WT + (size_t)(n0 + t * 16 + lm) * K + kk + hf * 16;
      v16h bf = *(const v16h*)wp;
      acc[t] = wmma_f16(af, bf, acc[t]);
    }
  }
#pragma unroll
  for (int t = 0; t < 4; t++) {
    int col = n0 + t * 16 + lm;
    float bv = bias[col];
#pragma unroll
    for (int r = 0; r < 8; r++) {
      int row = m0 + r + 8 * hf;
      float v = acc[t][r] + bv;
      if (ACT == 1) v = gelu_tanh(v);
      if (ADDF) outF[(size_t)row * N + col] += v;
      else      outH[(size_t)row * N + col] = (_Float16)v;
    }
  }
}

// ---------------------------------------------------------------------------
// Flash attention per (batch, head): T=256, dh=32, causal.
// 8 waves/block; wave w handles q-tiles w and w+8.  V^T staged in LDS,
// probabilities re-laid-out through LDS into A-fragments for P@V WMMA.
// ---------------------------------------------------------------------------
__global__ __launch_bounds__(256) void attn_kernel(
    const _Float16* __restrict__ Q, const _Float16* __restrict__ Kb,
    const _Float16* __restrict__ Vb, _Float16* __restrict__ O) {
  int b = blockIdx.x >> 2;
  int head = blockIdx.x & 3;
  __shared__ __align__(32) _Float16 VT[DH][Tt];      // 16 KB: V transposed (dh x T)
  __shared__ __align__(32) _Float16 P[8][16][32];    // 8 KB: per-wave prob staging
  int tid = threadIdx.x;
  for (int i = tid; i < DH * Tt; i += 256) {
    int d = i / Tt; int t = i % Tt;
    VT[d][t] = Vb[((size_t)b * Tt + t) * Hh + head * DH + d];
  }
  __syncthreads();
  int wave = tid >> 5, lane = tid & 31;
  int hf = lane >> 4, lm = lane & 15;
  const float scale = 0.17677669529663687f;  // 1/sqrt(32)

  for (int qt = wave; qt < 16; qt += 8) {
    // Q fragment (16 q-rows x 32 dh) straight from global (dh contiguous)
    v16h qa;
    {
      const _Float16* qp = Q + ((size_t)b * Tt + qt * 16 + lm) * Hh + head * DH + hf * 16;
      v8h lo = *(const v8h*)qp;
      v8h hi = *(const v8h*)(qp + 8);
#pragma unroll
      for (int i = 0; i < 8; i++) { qa[i] = lo[i]; qa[8 + i] = hi[i]; }
    }
    float m[8], l[8];
    v8f o0 = {}, o1 = {};
#pragma unroll
    for (int r = 0; r < 8; r++) { m[r] = -1e30f; l[r] = 0.f; }

    int npair = (qt >> 1) + 1;                 // 32-key chunks covering causal span
    for (int j = 0; j < npair; j++) {
      v8f s0 = {}, s1 = {};
      {  // logits: K^T fragments (lane=key column, 16 contiguous dh dims)
        const _Float16* kp0 = Kb + ((size_t)b * Tt + 32 * j + lm) * Hh + head * DH + hf * 16;
        v16h kb0 = *(const v16h*)kp0;
        s0 = wmma_f16(qa, kb0, s0);
        const _Float16* kp1 = Kb + ((size_t)b * Tt + 32 * j + 16 + lm) * Hh + head * DH + hf * 16;
        v16h kb1 = *(const v16h*)kp1;
        s1 = wmma_f16(qa, kb1, s1);
      }
      float pm[8];
#pragma unroll
      for (int r = 0; r < 8; r++) {
        int grow = qt * 16 + r + 8 * hf;
        float a0 = s0[r] * scale; if (32 * j + lm > grow)      a0 = -1e30f;
        float a1 = s1[r] * scale; if (32 * j + 16 + lm > grow) a1 = -1e30f;
        s0[r] = a0; s1[r] = a1;
        pm[r] = fmaxf(a0, a1);
      }
      for (int off = 1; off < 16; off <<= 1)
#pragma unroll
        for (int r = 0; r < 8; r++) pm[r] = fmaxf(pm[r], __shfl_xor(pm[r], off, 32));
      float alpha[8], rs[8];
#pragma unroll
      for (int r = 0; r < 8; r++) {
        float mn = fmaxf(m[r], pm[r]);
        alpha[r] = __expf(m[r] - mn);
        m[r] = mn;
        float p0 = __expf(s0[r] - mn);
        float p1 = __expf(s1[r] - mn);
        s0[r] = p0; s1[r] = p1;
        rs[r] = p0 + p1;
      }
      for (int off = 1; off < 16; off <<= 1)
#pragma unroll
        for (int r = 0; r < 8; r++) rs[r] += __shfl_xor(rs[r], off, 32);
#pragma unroll
      for (int r = 0; r < 8; r++) {
        l[r] = l[r] * alpha[r] + rs[r];
        o0[r] *= alpha[r];
        o1[r] *= alpha[r];
      }
      // stage probs (C layout) -> LDS -> reload as A fragment (in-order DS per wave)
#pragma unroll
      for (int r = 0; r < 8; r++) {
        P[wave][r + 8 * hf][lm]      = (_Float16)s0[r];
        P[wave][r + 8 * hf][16 + lm] = (_Float16)s1[r];
      }
      asm volatile("s_wait_dscnt 0" ::: "memory");
      v16h pa;
      {
        const _Float16* pp = &P[wave][lm][hf * 16];
        v8h lo = *(const v8h*)pp;
        v8h hi = *(const v8h*)(pp + 8);
#pragma unroll
        for (int i = 0; i < 8; i++) { pa[i] = lo[i]; pa[8 + i] = hi[i]; }
      }
      {  // P @ V: B fragments from transposed V in LDS (contiguous keys)
        const _Float16* vp0 = &VT[lm][32 * j + hf * 16];
        v16h vb0 = *(const v16h*)vp0;
        o0 = wmma_f16(pa, vb0, o0);
        const _Float16* vp1 = &VT[16 + lm][32 * j + hf * 16];
        v16h vb1 = *(const v16h*)vp1;
        o1 = wmma_f16(pa, vb1, o1);
      }
    }
#pragma unroll
    for (int r = 0; r < 8; r++) {
      float il = 1.0f / l[r];
      int grow = qt * 16 + r + 8 * hf;
      size_t base = ((size_t)b * Tt + grow) * Hh + head * DH;
      O[base + lm]      = (_Float16)(o0[r] * il);
      O[base + 16 + lm] = (_Float16)(o1[r] * il);
    }
  }
}

// ---------------------------------------------------------------------------
// GMM head: out[b,d] for d<64
// ---------------------------------------------------------------------------
__global__ __launch_bounds__(128) void cont_head_kernel(
    const float* __restrict__ h, const int* __restrict__ inv,
    const float* __restrict__ x, const float* __restrict__ gw,
    const float* __restrict__ gb, float* __restrict__ out) {
  int blk = blockIdx.x; int b = blk >> 6; int df = blk & 63;
  int t = threadIdx.x;
  int pos = 2 * inv[b * Dd + df];
  const float* hv = h + ((size_t)b * Tt + pos) * Hh;
  __shared__ float sh[128];
  __shared__ float g[304];
  __shared__ float ta[128], tb[128];
  sh[t] = hv[t];
  __syncthreads();
  for (int n = t; n < 300; n += 128) {
    float a = gb[n];
    for (int k = 0; k < 128; k++) a += sh[k] * gw[k * 300 + n];
    g[n] = a;
  }
  __syncthreads();
  float v = x[b * Dd + df];
  float term = -1e30f, mix = -1e30f;
  if (t < 100) {
    mix = g[t];
    float loc = g[100 + t], ls = g[200 + t];
    float z = (v - loc) * __expf(-ls);
    term = mix + (-0.5f * z * z - ls - 0.9189385332046727f);
  }
  ta[t] = term; tb[t] = mix; __syncthreads();
  for (int s = 64; s > 0; s >>= 1) {
    if (t < s) { ta[t] = fmaxf(ta[t], ta[t + s]); tb[t] = fmaxf(tb[t], tb[t + s]); }
    __syncthreads();
  }
  float mt = ta[0], mm = tb[0];
  __syncthreads();
  ta[t] = (t < 100) ? __expf(term - mt) : 0.f;
  tb[t] = (t < 100) ? __expf(mix - mm) : 0.f;
  __syncthreads();
  for (int s = 64; s > 0; s >>= 1) {
    if (t < s) { ta[t] += ta[t + s]; tb[t] += tb[t + s]; }
    __syncthreads();
  }
  if (t == 0) out[b * Dd + df] = (mt + __logf(ta[0])) - (mm + __logf(tb[0]));
}

// ---------------------------------------------------------------------------
// Categorical head: out[b, 64+kf]
// ---------------------------------------------------------------------------
__global__ __launch_bounds__(32) void disc_head_kernel(
    const float* __restrict__ h, const int* __restrict__ inv,
    const float* __restrict__ x, const float* __restrict__ cw,
    const float* __restrict__ cb, float* __restrict__ out) {
  int blk = blockIdx.x; int b = blk >> 6; int kf = blk & 63;
  int t = threadIdx.x;
  int d = 64 + kf;
  int pos = 2 * inv[b * Dd + d];
  const float* hv = h + ((size_t)b * Tt + pos) * Hh;
  __shared__ float lg[10];
  if (t < 10) {
    float a = cb[t];
    for (int k = 0; k < 128; k++) a += hv[k] * cw[k * 10 + t];
    lg[t] = a;
  }
  __syncthreads();
  if (t == 0) {
    int c = (int)x[b * Dd + d];
    float mx = -1e30f;
    for (int i = 0; i < 10; i++) mx = fmaxf(mx, lg[i]);
    float s = 0.f;
    for (int i = 0; i < 10; i++) s += __expf(lg[i] - mx);
    out[b * Dd + d] = lg[c] - (mx + __logf(s));
  }
}

// ---------------------------------------------------------------------------
// Host orchestration
// ---------------------------------------------------------------------------
extern "C" void kernel_launch(void* const* d_in, const int* in_sizes, int n_in,
                              void* d_out, int out_size, void* d_ws, size_t ws_size,
                              hipStream_t stream) {
  (void)in_sizes; (void)n_in; (void)out_size; (void)ws_size;
  char* ws = (char*)d_ws;

  // input indices (setup_inputs dict order, depth-first)
  const float* x        = (const float*)d_in[0];
  const int*   order    = (const int*)d_in[1];
  const float* iemb     = (const float*)d_in[2];
  const float* cemb     = (const float*)d_in[3];
  // 4..7 index_mlp, 8..11 cont_mlp, 12..15 disc_mlp, 16+16L layers, 80..83 heads
  auto F = [&](int i) { return (const float*)d_in[i]; };

  // workspace layout (bytes)
  const size_t OFF_WT    = 0;          // f16 transposed weights (1.57 MB)
  const size_t OFF_INV   = 1572864;    // int inv (128 KB)
  const size_t OFF_IDXT  = 1703936;    // 128x128 f32
  const size_t OFF_DISCT = 1769472;    // 640x128 f32
  const size_t OFF_CONTF = 2097152;    // 256*64*128 f32 (8.4 MB)
  const size_t OFF_H     = 10485760;   // residual f32 (33.5 MB)
  const size_t OFF_XLN   = 44040192;   // ln out f16 (16.8 MB)
  const size_t OFF_Q     = 60817408;   // q f16; also FFN intermediate start
  const size_t OFF_K     = 77594624;
  const size_t OFF_V     = 94371840;
  const size_t OFF_O     = 111149056;

  _Float16* WT    = (_Float16*)(ws + OFF_WT);
  int*      inv   = (int*)(ws + OFF_INV);
  float*    idxT  = (float*)(ws + OFF_IDXT);
  float*    discT = (float*)(ws + OFF_DISCT);
  float*    contf = (float*)(ws + OFF_CONTF);
  float*    h     = (float*)(ws + OFF_H);
  _Float16* xln   = (_Float16*)(ws + OFF_XLN);
  _Float16* qb    = (_Float16*)(ws + OFF_Q);
  _Float16* kb    = (_Float16*)(ws + OFF_K);
  _Float16* vb    = (_Float16*)(ws + OFF_V);
  _Float16* ob    = (_Float16*)(ws + OFF_O);
  _Float16* ffn   = (_Float16*)(ws + OFF_Q);  // reuses q/k/v/o region (67.1 MB)

  const size_t LWT = 196608;  // halves per layer of transposed weights
  // 1) convert weights to f16 transposed
  for (int L = 0; L < 4; L++) {
    int bi = 16 + 16 * L;
    _Float16* lw = WT + (size_t)L * LWT;
    convT_kernel<<<(16384 + 255) / 256, 256, 0, stream>>>(F(bi + 2),  lw + 0,      128, 128); // wq
    convT_kernel<<<(16384 + 255) / 256, 256, 0, stream>>>(F(bi + 4),  lw + 16384,  128, 128); // wk
    convT_kernel<<<(16384 + 255) / 256, 256, 0, stream>>>(F(bi + 6),  lw + 32768,  128, 128); // wv
    convT_kernel<<<(16384 + 255) / 256, 256, 0, stream>>>(F(bi + 8),  lw + 49152,  128, 128); // wo
    convT_kernel<<<(65536 + 255) / 256, 256, 0, stream>>>(F(bi + 12), lw + 65536,  128, 512); // fc1
    convT_kernel<<<(65536 + 255) / 256, 256, 0, stream>>>(F(bi + 14), lw + 131072, 512, 128); // fc2
  }

  // 2) permutation inverse + feature tables + per-sample cont features
  inv_kernel<<<Bsz, Dd, 0, stream>>>(order, inv);
  idx_table_kernel<<<128, 128, 0, stream>>>(iemb, F(4), F(5), F(6), F(7), idxT);
  disc_table_kernel<<<640, 128, 0, stream>>>(iemb, cemb, F(12), F(13), F(14), F(15), discT);
  cont_feat_kernel<<<Bsz * 64, 128, 0, stream>>>(iemb, x, F(8), F(9), F(10), F(11), contf);
  assemble_kernel<<<Bsz * Dd, 128, 0, stream>>>(order, x, idxT, discT, contf, h);

  // 3) transformer blocks
  const int Mtok = Bsz * Tt;           // 65536 tokens
  dim3 g128(Mtok / 16, 128 / 64);      // (4096, 2)
  dim3 g512(Mtok / 16, 512 / 64);      // (4096, 8)
  for (int L = 0; L < 4; L++) {
    int bi = 16 + 16 * L;
    _Float16* lw = WT + (size_t)L * LWT;
    ln_kernel<<<Mtok, 128, 0, stream>>>(h, F(bi + 0), F(bi + 1), xln);
    gemm_wmma_kernel<0, 0><<<g128, 32, 0, stream>>>(xln, lw + 0,     F(bi + 3), qb, nullptr, 128, 128);
    gemm_wmma_kernel<0, 0><<<g128, 32, 0, stream>>>(xln, lw + 16384, F(bi + 5), kb, nullptr, 128, 128);
    gemm_wmma_kernel<0, 0><<<g128, 32, 0, stream>>>(xln, lw + 32768, F(bi + 7), vb, nullptr, 128, 128);
    attn_kernel<<<Bsz * NHh, 256, 0, stream>>>(qb, kb, vb, ob);
    gemm_wmma_kernel<0, 1><<<g128, 32, 0, stream>>>(ob, lw + 49152, F(bi + 9), nullptr, h, 128, 128);
    ln_kernel<<<Mtok, 128, 0, stream>>>(h, F(bi + 10), F(bi + 11), xln);
    gemm_wmma_kernel<1, 0><<<g512, 32, 0, stream>>>(xln, lw + 65536,  F(bi + 13), ffn, nullptr, 128, 512);
    gemm_wmma_kernel<0, 1><<<g128, 32, 0, stream>>>(ffn, lw + 131072, F(bi + 15), nullptr, h, 512, 128);
  }

  // 4) likelihood heads -> d_out (B x D f32)
  cont_head_kernel<<<Bsz * 64, 128, 0, stream>>>(h, inv, x, F(80), F(81), (float*)d_out);
  disc_head_kernel<<<Bsz * 64, 32, 0, stream>>>(h, inv, x, F(82), F(83), (float*)d_out);
}
